// GPT2Attention_45200235823410
// MI455X (gfx1250) — matrix operand order repaired and checked by reading, not compiled, not run on hardware
//
#include <hip/hip_runtime.h>

#define E     768
#define NH    12
#define HD    64
#define BB    2
#define TT    2048
#define MROWS (BB*TT)   // 4096
#define QKV   (3*E)     // 2304

typedef __attribute__((ext_vector_type(16))) _Float16 v16h;
typedef __attribute__((ext_vector_type(8)))  float    v8f;
typedef unsigned int u32x4 __attribute__((ext_vector_type(4)));
typedef int          i32x4 __attribute__((ext_vector_type(4)));
typedef int          i32x8 __attribute__((ext_vector_type(8)));

union F16x16 { v16h h; unsigned int u[8]; };

#ifndef __has_builtin
#define __has_builtin(x) 0
#endif
#if defined(__AMDGCN__) && __has_builtin(__builtin_amdgcn_tensor_load_to_lds) && \
    __has_builtin(__builtin_amdgcn_s_wait_tensorcnt)
#define USE_TDM 1
#else
#define USE_TDM 0
#endif
#if __has_include(<hip/amd_detail/amd_gfx1250_TDM.h>)
#define TDM_SIX_ARGS 1
#else
#define TDM_SIX_ARGS 0
#endif

__device__ __forceinline__ v8f wmma16(v16h a, v16h b, v8f c) {
  return __builtin_amdgcn_wmma_f32_16x16x32_f16(false, a, false, b, (short)0, c, false, false);
}

// A fragment (16x32 f16): lane m = row, halves offset K by 8; VGPR p -> K = 2p (+8 if p>=4).
__device__ __forceinline__ v16h load_a_frag(const _Float16* base, int ldk, int m, int half) {
  F16x16 f;
  const _Float16* row = base + m * ldk + half * 8;
  #pragma unroll
  for (int p = 0; p < 8; ++p) {
    int k = 2 * p + ((p >= 4) ? 8 : 0);
    f.u[p] = *(const unsigned int*)(row + k);
  }
  return f.h;
}

// B fragment (32x16 f16): caller points at K=0 of column n (K contiguous);
// halves offset K by 8; VGPR p -> K = 2p (+8 if p>=4), same as A.
__device__ __forceinline__ v16h load_b_frag(const _Float16* col, int half) {
  F16x16 f;
  const _Float16* p0 = col + half * 8;
  #pragma unroll
  for (int p = 0; p < 8; ++p) f.u[p] = *(const unsigned int*)(p0 + 2 * p + ((p >= 4) ? 8 : 0));
  return f.h;
}

// ---- TDM: async DMA of one 128-row x 32-elem f16 tile (row pitch = stride) ----
__device__ __forceinline__ void tdm_load_tile(_Float16* lds, const _Float16* gsrc,
                                              unsigned tdim0, unsigned tdim1,
                                              unsigned stride, int lane) {
#if USE_TDM
  unsigned lds_off = (unsigned)(uintptr_t)lds;        // low 32 bits of flat = LDS offset
  unsigned long long ga = (unsigned long long)(uintptr_t)gsrc;
  u32x4 g0;
  g0[0] = 1u;                                          // count=1, user descriptor
  g0[1] = lds_off;                                     // lds_addr [63:32]
  g0[2] = (unsigned)ga;                                // global_addr [95:64]
  g0[3] = (unsigned)((ga >> 32) & 0x01ffffffu) | 0x80000000u; // addr[56:32] | type=2
  i32x8 g1;
  g1[0] = 0x00010000;                                  // data_size = 2 bytes
  g1[1] = (int)((tdim0 & 0xffffu) << 16);              // tensor_dim0[15:0] @63:48
  g1[2] = (int)((tdim0 >> 16) | ((tdim1 & 0xffffu) << 16));   // dim0 hi | dim1 lo
  g1[3] = (int)((tdim1 >> 16) | (32u << 16));          // dim1 hi | tile_dim0=32
  g1[4] = 128;                                         // tile_dim1=128, tile_dim2=0
  g1[5] = (int)stride;                                 // tensor_dim0_stride[31:0]
  g1[6] = 0;
  g1[7] = 0;
  i32x4 z4 = {0, 0, 0, 0};
#if TDM_SIX_ARGS
  i32x8 z8 = {0, 0, 0, 0, 0, 0, 0, 0};
  __builtin_amdgcn_tensor_load_to_lds(g0, g1, z4, z4, z8, 0);
#else
  __builtin_amdgcn_tensor_load_to_lds(g0, g1, z4, z4, 0);
#endif
#else
  // Fallback: wave-0 lane-parallel synchronous copy (16 dwords per 32-elem row).
  unsigned* d = (unsigned*)lds;
  for (int i = lane; i < 128 * 16; i += 32) {
    int r = i >> 4, c = i & 15;
    d[i] = *(const unsigned*)(gsrc + (size_t)r * stride + c * 2);
  }
#endif
}

__device__ __forceinline__ void tdm_wait0() {
#if USE_TDM
  __builtin_amdgcn_s_wait_tensorcnt(0);
#endif
}

// ---- Shared double-buffered WMMA GEMM main loop (128x128 block, 8 waves) ----
__device__ __forceinline__ void gemm_loop(const _Float16* __restrict__ Ag,  // block row origin
                                          const _Float16* __restrict__ Bg,  // block col origin (B^T row)
                                          int Kdim, unsigned tensM, unsigned tensN,
                                          _Float16* As, _Float16* Bs,       // [2][128*32] each
                                          int tid, v8f acc[4][2]) {
  const int lane = tid & 31, wave = tid >> 5;
  const int wm = wave >> 2, wn = wave & 3;     // 2 x 4 wave grid
  const int m = lane & 15, half = lane >> 4;
  const int nk = Kdim / 32;
  if (wave == 0) {
    tdm_load_tile(As, Ag, (unsigned)Kdim, tensM, (unsigned)Kdim, lane);
    tdm_load_tile(Bs, Bg, (unsigned)Kdim, tensN, (unsigned)Kdim, lane);
  }
  for (int kt = 0; kt < nk; ++kt) {
    const int cur = kt & 1, nxt = cur ^ 1;
    if (wave == 0) tdm_wait0();          // tile kt resident in LDS
    __syncthreads();                     // publish to all waves; frees buf 'nxt'
    if (wave == 0 && kt + 1 < nk) {      // overlap DMA of tile kt+1 with compute
      tdm_load_tile(As + nxt * 4096, Ag + (kt + 1) * 32, (unsigned)Kdim, tensM, (unsigned)Kdim, lane);
      tdm_load_tile(Bs + nxt * 4096, Bg + (kt + 1) * 32, (unsigned)Kdim, tensN, (unsigned)Kdim, lane);
    }
    const _Float16* Ab = As + cur * 4096;
    const _Float16* Bb = Bs + cur * 4096;
    v16h af[4], bf[2];
    #pragma unroll
    for (int i = 0; i < 4; ++i) af[i] = load_a_frag(Ab + (wm * 64 + i * 16) * 32, 32, m, half);
    #pragma unroll
    for (int j = 0; j < 2; ++j) bf[j] = load_b_frag(Bb + (wn * 32 + j * 16 + m) * 32, half);
    #pragma unroll
    for (int i = 0; i < 4; ++i)
      #pragma unroll
      for (int j = 0; j < 2; ++j)
        acc[i][j] = wmma16(af[i], bf[j], acc[i][j]);
  }
}

// ---------------- prep: f32 -> f16 copy and transposing convert --------------
__global__ void __launch_bounds__(256)
cvt_kernel(const float* __restrict__ s, _Float16* __restrict__ d, int n) {
  int i = blockIdx.x * blockDim.x + threadIdx.x;
  int stride = gridDim.x * blockDim.x;
  for (; i < n; i += stride) d[i] = (_Float16)s[i];
}

__global__ void __launch_bounds__(256)
transpose_cvt_kernel(const float* __restrict__ s, _Float16* __restrict__ d, int K, int N) {
  // d[n*K + k] = (f16)s[k*N + n]; block (32,8), 32x32 tile
  __shared__ _Float16 t[32][33];
  const int n0 = blockIdx.x * 32, k0 = blockIdx.y * 32;
  const int tx = threadIdx.x, ty = threadIdx.y;
  #pragma unroll
  for (int r = 0; r < 4; ++r)
    t[ty + r * 8][tx] = (_Float16)s[(size_t)(k0 + ty + r * 8) * N + n0 + tx];
  __syncthreads();
  #pragma unroll
  for (int r = 0; r < 4; ++r)
    d[(size_t)(n0 + ty + r * 8) * K + k0 + tx] = t[tx][ty + r * 8];
}

// ---------------- Kernel 1: QKV GEMM + fused head split ---------------------
// Q,K -> [b,n,t,h] f16 ; V -> transposed [b,n,h,t] f16. channel = h*NH + n.
__global__ void __launch_bounds__(256)
qkv_kernel(const _Float16* __restrict__ xh, const _Float16* __restrict__ waT,
           const float* __restrict__ bias,
           _Float16* __restrict__ Qb, _Float16* __restrict__ Kb,
           _Float16* __restrict__ Vt) {
  __shared__ _Float16 As[2 * 128 * 32];
  __shared__ _Float16 Bs[2 * 128 * 32];
  const int tid = threadIdx.x;
  const int lane = tid & 31, wave = tid >> 5;
  const int wm = wave >> 2, wn = wave & 3;
  const int m = lane & 15, half = lane >> 4;
  const int row0 = blockIdx.y * 128, col0 = blockIdx.x * 128;

  v8f acc[4][2] = {};
  gemm_loop(xh + (size_t)row0 * E, waT + (size_t)col0 * E, E, MROWS, QKV, As, Bs, tid, acc);

  #pragma unroll
  for (int j = 0; j < 2; ++j) {
    int gn = col0 + wn * 32 + j * 16 + m;           // [0, 2304)
    float bv = bias[gn];
    int mat = gn / E;                                // 0=Q 1=K 2=V
    int cc  = gn - mat * E;
    int h   = cc / NH;                               // head-dim (h is OUTER in channel)
    int hd  = cc - h * NH;                           // head index (INNER factor)
    #pragma unroll
    for (int i = 0; i < 4; ++i) {
      #pragma unroll
      for (int v = 0; v < 8; ++v) {
        int gm = row0 + wm * 64 + i * 16 + v + half * 8;
        int bi = gm >> 11;
        int t  = gm & (TT - 1);
        _Float16 hv = (_Float16)(acc[i][j][v] + bv);
        if (mat == 0)      Qb[((size_t)(bi * NH + hd) * TT + t) * HD + h] = hv;
        else if (mat == 1) Kb[((size_t)(bi * NH + hd) * TT + t) * HD + h] = hv;
        else               Vt[((size_t)(bi * NH + hd) * HD + h) * TT + t] = hv;
      }
    }
  }
}

// ---------------- Kernel 2: flash attention, one wave per 16-query tile -----
__global__ void __launch_bounds__(128)
attn_kernel(const _Float16* __restrict__ Qb, const _Float16* __restrict__ Kb,
            const _Float16* __restrict__ Vt, _Float16* __restrict__ att) {
  __shared__ _Float16 Pl[4][16 * 32];
  const int lane = threadIdx.x & 31, wave = threadIdx.x >> 5;
  const int m = lane & 15, half = lane >> 4;
  const int q0 = (blockIdx.x * 4 + wave) * 16;
  const int head = blockIdx.y;
  const int bi = head / NH, nh = head - bi * NH;
  const _Float16* Qh = Qb + (size_t)head * TT * HD;
  const _Float16* Kh = Kb + (size_t)head * TT * HD;
  const _Float16* Vh = Vt + (size_t)head * HD * TT;   // [h][s]
  _Float16* P = &Pl[wave][0];

  v16h aq[2];
  #pragma unroll
  for (int ks = 0; ks < 2; ++ks)
    aq[ks] = load_a_frag(Qh + (size_t)q0 * HD + ks * 32, HD, m, half);

  v8f o[4] = {};
  float mr[8], lr[8];
  #pragma unroll
  for (int v = 0; v < 8; ++v) { mr[v] = -3.0e38f; lr[v] = 0.0f; }

  const float scale = 0.125f;
  const int smax = q0 + 16;
  for (int s0 = 0; s0 < smax; s0 += 32) {
    v8f sc[2] = {};
    #pragma unroll
    for (int nt = 0; nt < 2; ++nt)
      #pragma unroll
      for (int ks = 0; ks < 2; ++ks) {
        v16h bk = load_b_frag(Kh + (size_t)(s0 + nt * 16 + m) * HD + ks * 32, half);
        sc[nt] = wmma16(aq[ks], bk, sc[nt]);
      }
    float cmax[8];
    #pragma unroll
    for (int v = 0; v < 8; ++v) {
      int r = q0 + v + half * 8;
      float a0 = sc[0][v] * scale;
      float a1 = sc[1][v] * scale;
      if (s0 + m > r)      a0 = -3.0e38f;
      if (s0 + 16 + m > r) a1 = -3.0e38f;
      sc[0][v] = a0; sc[1][v] = a1;
      cmax[v] = fmaxf(a0, a1);
    }
    #pragma unroll
    for (int d = 1; d < 16; d <<= 1)
      #pragma unroll
      for (int v = 0; v < 8; ++v)
        cmax[v] = fmaxf(cmax[v], __shfl_xor(cmax[v], d, 32));
    float alpha[8], rs[8];
    #pragma unroll
    for (int v = 0; v < 8; ++v) {
      float mn = fmaxf(mr[v], cmax[v]);
      alpha[v] = __expf(mr[v] - mn);
      mr[v] = mn;
      float p0 = __expf(sc[0][v] - mn);
      float p1 = __expf(sc[1][v] - mn);
      sc[0][v] = p0; sc[1][v] = p1;
      rs[v] = p0 + p1;
    }
    #pragma unroll
    for (int d = 1; d < 16; d <<= 1)
      #pragma unroll
      for (int v = 0; v < 8; ++v)
        rs[v] += __shfl_xor(rs[v], d, 32);
    #pragma unroll
    for (int v = 0; v < 8; ++v) lr[v] = lr[v] * alpha[v] + rs[v];
    #pragma unroll
    for (int j = 0; j < 4; ++j)
      #pragma unroll
      for (int v = 0; v < 8; ++v)
        o[j][v] *= alpha[v];
    #pragma unroll
    for (int v = 0; v < 8; ++v) {
      P[(v + half * 8) * 32 + m]      = (_Float16)sc[0][v];
      P[(v + half * 8) * 32 + 16 + m] = (_Float16)sc[1][v];
    }
    asm volatile("s_wait_dscnt 0x0" ::: "memory");
    v16h pf = load_a_frag(P, 32, m, half);
    #pragma unroll
    for (int j = 0; j < 4; ++j) {
      v16h bv = load_b_frag(Vh + (size_t)(j * 16 + m) * TT + s0, half);
      o[j] = wmma16(pf, bv, o[j]);
    }
  }
  #pragma unroll
  for (int v = 0; v < 8; ++v) {
    float inv = 1.0f / lr[v];
    int t = q0 + v + half * 8;
    #pragma unroll
    for (int j = 0; j < 4; ++j) {
      int h = j * 16 + m;
      att[((size_t)(bi * TT + t)) * E + nh * HD + h] = (_Float16)(o[j][v] * inv);
    }
  }
}

// ---------------- Kernel 3: out = att @ w_proj + b_proj (f32 out) ----------
__global__ void __launch_bounds__(256)
proj_kernel(const _Float16* __restrict__ att, const _Float16* __restrict__ wpT,
            const float* __restrict__ bias, float* __restrict__ out) {
  __shared__ _Float16 As[2 * 128 * 32];
  __shared__ _Float16 Bs[2 * 128 * 32];
  const int tid = threadIdx.x;
  const int lane = tid & 31, wave = tid >> 5;
  const int wm = wave >> 2, wn = wave & 3;
  const int m = lane & 15, half = lane >> 4;
  const int row0 = blockIdx.y * 128, col0 = blockIdx.x * 128;

  v8f acc[4][2] = {};
  gemm_loop(att + (size_t)row0 * E, wpT + (size_t)col0 * E, E, MROWS, E, As, Bs, tid, acc);

  #pragma unroll
  for (int j = 0; j < 2; ++j) {
    int gn = col0 + wn * 32 + j * 16 + m;
    float bv = bias[gn];
    #pragma unroll
    for (int i = 0; i < 4; ++i)
      #pragma unroll
      for (int v = 0; v < 8; ++v) {
        int gm = row0 + wm * 64 + i * 16 + v + half * 8;
        out[(size_t)gm * E + gn] = acc[i][j][v] + bv;
      }
  }
}

extern "C" void kernel_launch(void* const* d_in, const int* in_sizes, int n_in,
                              void* d_out, int out_size, void* d_ws, size_t ws_size,
                              hipStream_t stream) {
  const float* x      = (const float*)d_in[0];
  const float* w_attn = (const float*)d_in[1];
  const float* b_attn = (const float*)d_in[2];
  const float* w_proj = (const float*)d_in[3];
  const float* b_proj = (const float*)d_in[4];
  float* out = (float*)d_out;

  _Float16* p = (_Float16*)d_ws;
  _Float16* xh  = p; p += (size_t)MROWS * E;        // 4096*768
  _Float16* waT = p; p += (size_t)QKV * E;          // 2304*768 (transposed)
  _Float16* wpT = p; p += (size_t)E * E;            // 768*768  (transposed)
  const size_t per = (size_t)BB * NH * TT * HD;
  _Float16* Qb  = p; p += per;
  _Float16* Kb  = p; p += per;
  _Float16* Vt  = p; p += per;
  _Float16* att = p;                                 // total ~36 MB

  cvt_kernel<<<2048, 256, 0, stream>>>(x, xh, MROWS * E);
  transpose_cvt_kernel<<<dim3(QKV / 32, E / 32), dim3(32, 8), 0, stream>>>(w_attn, waT, E, QKV);
  transpose_cvt_kernel<<<dim3(E / 32,  E / 32), dim3(32, 8), 0, stream>>>(w_proj, wpT, E, E);

  qkv_kernel <<<dim3(QKV / 128, MROWS / 128), 256, 0, stream>>>(xh, waT, b_attn, Qb, Kb, Vt);
  attn_kernel<<<dim3((TT / 16) / 4, BB * NH),  128, 0, stream>>>(Qb, Kb, Vt, att);
  proj_kernel<<<dim3(E / 128, MROWS / 128),    256, 0, stream>>>(att, wpT, b_proj, out);
}
